// TemporalAttention_1314259992602
// MI455X (gfx1250) — matrix operand order repaired and checked by reading, not compiled, not run on hardware
//
#include <hip/hip_runtime.h>
#include <cstdint>
#include <cstddef>

#define BATCH 64
#define SEQ   4096
#define HID   256   // H == K == 256

typedef __attribute__((ext_vector_type(16))) __bf16 v16bf;
typedef __attribute__((ext_vector_type(8)))  float  v8f;
typedef unsigned int u32x4 __attribute__((ext_vector_type(4)));
typedef int          i32x8 __attribute__((ext_vector_type(8)));
typedef int          i32x4 __attribute__((ext_vector_type(4)));

// Branch-free tanh: 1 - 2/(e^{2x}+1). exp->inf => +1, exp->0 => -1. No EXEC games.
static __device__ __forceinline__ float fast_tanh(float x) {
    float e = __expf(2.0f * x);
    return 1.0f - 2.0f * __builtin_amdgcn_rcpf(e + 1.0f);
}

// ---------------------------------------------------------------------------
// TDM: DMA one 16x256 bf16 tile (8KB) from global into LDS.
// D# per CDNA5 ISA 8.3/8.4: count=1, type=2(image), data_size=1(2B),
// 2D: tile_dim0=256, tile_dim1=16, tensor_dim0=256, tensor_dim1=16,
// tensor_dim0_stride=256. Groups 2/3 zero (<=2D tensor).
// ---------------------------------------------------------------------------
static __device__ __forceinline__ void tdm_load_tile_16x256(const void* gptr,
                                                            unsigned lds_addr) {
    unsigned long long ga = (unsigned long long)(uintptr_t)gptr;
    u32x4 g0;
    g0[0] = 1u;                                                  // count=1
    g0[1] = lds_addr;                                            // lds_addr (bytes)
    g0[2] = (unsigned)ga;                                        // global_addr[31:0]
    g0[3] = (unsigned)((ga >> 32) & 0x1FFFFFFull) | (2u << 30);  // ga[56:32] | type=2
    i32x8 g1;
    g1[0] = 0x00010000;             // workgroup_mask=0, data_size=1 (2 bytes)
    g1[1] = (int)(256u << 16);      // tensor_dim0[15:0]=256   (bits 63:48)
    g1[2] = (int)(16u << 16);       // tensor_dim1[15:0]=16    (bits 95:80)
    g1[3] = (int)(256u << 16);      // tile_dim0=256           (bits 127:112)
    g1[4] = 16;                     // tile_dim1=16            (bits 143:128)
    g1[5] = 256;                    // tensor_dim0_stride=256  (bits 191:160)
    g1[6] = 0;
    g1[7] = 0;
    i32x4 z4 = {0, 0, 0, 0};
    i32x8 z8 = {0, 0, 0, 0, 0, 0, 0, 0};
    // 6-arg form on this toolchain: (g0, g1, g2, g3, extra, cpol)
    __builtin_amdgcn_tensor_load_to_lds(g0, g1, z4, z4, z8, 0);
}

// ---------------------------------------------------------------------------
// Phase 0: W1 fp32 -> bf16 once (it is re-read by all 2048 phase-2 blocks).
// grid 64, block 256; 4 elements/thread.
// ---------------------------------------------------------------------------
__global__ __launch_bounds__(256) void ta_phase0_w1bf(
    const float* __restrict__ W1, __bf16* __restrict__ W1bf) {
    const int i = blockIdx.x * 1024 + threadIdx.x * 4;
    float4 w = *(const float4*)(W1 + i);
    union { __bf16 e[4]; unsigned long long u; } p;
    p.e[0] = (__bf16)w.x; p.e[1] = (__bf16)w.y;
    p.e[2] = (__bf16)w.z; p.e[3] = (__bf16)w.w;
    *(unsigned long long*)(W1bf + i) = p.u;
}

// ---------------------------------------------------------------------------
// Phase 1: h2[b,k] = sum_h final_h[b,h] * W2[k,h]      (grid 64, block 256)
// ---------------------------------------------------------------------------
__global__ __launch_bounds__(256) void ta_phase1_h2(
    const float* __restrict__ final_h, const float* __restrict__ W2,
    float* __restrict__ h2) {
    const int b = blockIdx.x;
    const int k = threadIdx.x;
    const float4* fh = (const float4*)(final_h + (size_t)b * HID);
    const float4* w  = (const float4*)(W2 + (size_t)k * HID);
    float acc = 0.f;
#pragma unroll 8
    for (int i = 0; i < HID / 4; ++i) {
        float4 a = fh[i], c = w[i];
        acc += a.x * c.x + a.y * c.y + a.z * c.z + a.w * c.w;
    }
    h2[(size_t)b * HID + k] = acc;
}

// ---------------------------------------------------------------------------
// Phase 2: scores[b,s] = sum_k tanh( enc[b,s,:]·W1[k,:] + h2[b,k] ) * v[k]
// Block = 256 threads = 8 waves; each wave owns a 16-row s-tile (same b per
// block). W1bf k-tiles are DMA'd into double-buffered LDS by the TDM while
// the previous tile is consumed by WMMAs.
// ---------------------------------------------------------------------------
__global__ __launch_bounds__(256) void ta_phase2_scores(
    const float* __restrict__ enc, const __bf16* __restrict__ W1bf,
    const float* __restrict__ h2,  const float* __restrict__ v,
    float* __restrict__ scores) {

    __shared__ __align__(32) unsigned short Btile[2][16 * HID]; // bf16 bits, [n][h]

    const int t    = threadIdx.x;
    const int wave = t >> 5;
    const int lane = t & 31;
    const int tile = blockIdx.x * 8 + wave;     // 0..16383
    const int b    = tile >> 8;                 // S/16 == 256 tiles per batch
    const int s0   = (tile & 255) << 4;
    const int m    = lane & 15;                 // A row within tile
    const int half = lane >> 4;                 // lane-half selector
    const int n    = lane & 15;                 // output column within k-tile

    const unsigned ldsBase = (unsigned)(uintptr_t)&Btile[0][0];

    // ---- Prolog: DMA tile 0 into buffer 0.
    if (wave == 0) {
        tdm_load_tile_16x256(W1bf, ldsBase);
    }

    // ---- Load A: 16 rows x 256 cols of enc, once, as bf16 in WMMA A layout.
    // 16-bit A 16x32: slots 0..15 <- K0..K0+15,
    // K0 = 32*chunk + 16*half.  (overlaps the tile-0 DMA)
    const float* rowp = enc + ((size_t)b * SEQ + (size_t)(s0 + m)) * HID;
    v16bf A[8];
#pragma unroll
    for (int c = 0; c < 8; ++c) {
        const int K0 = 32 * c + 16 * half;
        float4 f0 = *(const float4*)(rowp + K0);
        float4 f1 = *(const float4*)(rowp + K0 + 4);
        float4 f2 = *(const float4*)(rowp + K0 + 8);
        float4 f3 = *(const float4*)(rowp + K0 + 12);
        union { v16bf vv; __bf16 e[16]; } av;
        av.e[0]  = (__bf16)f0.x; av.e[1]  = (__bf16)f0.y;
        av.e[2]  = (__bf16)f0.z; av.e[3]  = (__bf16)f0.w;
        av.e[4]  = (__bf16)f1.x; av.e[5]  = (__bf16)f1.y;
        av.e[6]  = (__bf16)f1.z; av.e[7]  = (__bf16)f1.w;
        av.e[8]  = (__bf16)f2.x; av.e[9]  = (__bf16)f2.y;
        av.e[10] = (__bf16)f2.z; av.e[11] = (__bf16)f2.w;
        av.e[12] = (__bf16)f3.x; av.e[13] = (__bf16)f3.y;
        av.e[14] = (__bf16)f3.z; av.e[15] = (__bf16)f3.w;
        A[c] = av.vv;
    }

    // ---- Hoist all per-lane bias (h2) and v values out of the k-tile loop.
    float biasv[16], vv[16];
#pragma unroll
    for (int i = 0; i < 16; ++i) {
        biasv[i] = h2[(size_t)b * HID + i * 16 + n];
        vv[i]    = v[i * 16 + n];
    }

    float sacc[8];
#pragma unroll
    for (int r = 0; r < 8; ++r) sacc[r] = 0.f;

    // Publish tile 0.
    if (wave == 0) __builtin_amdgcn_s_wait_tensorcnt(0);
    __syncthreads();

#pragma unroll
    for (int n0 = 0; n0 < 16; ++n0) {
        const unsigned short* bt = &Btile[n0 & 1][0];

        // ---- Kick DMA of the next tile into the other buffer (overlaps WMMAs).
        if (wave == 0 && n0 < 15)
            tdm_load_tile_16x256(W1bf + (size_t)(n0 + 1) * 16 * HID,
                                 ldsBase + (unsigned)(((n0 + 1) & 1) * 16 * HID * 2));

        // ---- C initialized with broadcast bias h2[b, n0*16+n]
        v8f c;
#pragma unroll
        for (int r = 0; r < 8; ++r) c[r] = biasv[n0];

        // ---- 8 WMMAs over H; batch 4 B-fragment loads per wait.
        // B uses the same contiguous k order as A: lane-half holds 16 contiguous K values.
#pragma unroll
        for (int g = 0; g < 2; ++g) {
            v16bf bf[4];
#pragma unroll
            for (int j = 0; j < 4; ++j)
                bf[j] = *(const v16bf*)&bt[n * HID + 32 * (4 * g + j) + 16 * half];
#pragma unroll
            for (int j = 0; j < 4; ++j)
                c = __builtin_amdgcn_wmma_f32_16x16x32_bf16(
                        false, A[4 * g + j], false, bf[j], (short)0, c, false, false);
        }

        // ---- tanh + dot with v, accumulate partial scores (row M = r + 8*half)
#pragma unroll
        for (int r = 0; r < 8; ++r)
            sacc[r] += fast_tanh(c[r]) * vv[n0];

        // ---- Publish next tile; barrier also guards buffer reuse.
        if (wave == 0 && n0 < 15) __builtin_amdgcn_s_wait_tensorcnt(0);
        __syncthreads();
    }

    // ---- Reduce across the 16 lanes of each half (xor masks stay in-group)
#pragma unroll
    for (int off = 8; off >= 1; off >>= 1)
#pragma unroll
        for (int r = 0; r < 8; ++r)
            sacc[r] += __shfl_xor(sacc[r], off, 32);

    if ((lane & 15) == 0) {
#pragma unroll
        for (int r = 0; r < 8; ++r)
            scores[(size_t)b * SEQ + s0 + r + 8 * half] = sacc[r];
    }
}

// ---------------------------------------------------------------------------
// Phase 3a: alpha[b,:] = softmax(scores[b,:])           (grid 64, block 256)
// ---------------------------------------------------------------------------
__global__ __launch_bounds__(256) void ta_phase3_softmax(
    const float* __restrict__ scores, float* __restrict__ alpha) {
    __shared__ float red[256];
    const int b = blockIdx.x, t = threadIdx.x;
    const float* sc = scores + (size_t)b * SEQ;

    float mx = -3.0e38f;
    for (int s = t; s < SEQ; s += 256) mx = fmaxf(mx, sc[s]);
    red[t] = mx; __syncthreads();
    for (int off = 128; off >= 1; off >>= 1) {
        if (t < off) red[t] = fmaxf(red[t], red[t + off]);
        __syncthreads();
    }
    mx = red[0]; __syncthreads();

    float sum = 0.f;
    for (int s = t; s < SEQ; s += 256) sum += __expf(sc[s] - mx);
    red[t] = sum; __syncthreads();
    for (int off = 128; off >= 1; off >>= 1) {
        if (t < off) red[t] += red[t + off];
        __syncthreads();
    }
    const float inv = 1.f / red[0];

    for (int s = t; s < SEQ; s += 256)
        alpha[(size_t)b * SEQ + s] = __expf(sc[s] - mx) * inv;
}

// ---------------------------------------------------------------------------
// Phase 3b: partial[b,chunk,h] = sum_{s in chunk} alpha[b,s]*enc[b,s,h]
// grid 64*16, block 256 (thread == h). Block index REVERSED so the re-read of
// enc starts with the most-recently-touched lines from phase 2 (192MB L2).
// enc reads are last-use -> non-temporal.
// ---------------------------------------------------------------------------
__global__ __launch_bounds__(256) void ta_phase3_context_partial(
    const float* __restrict__ enc, const float* __restrict__ alpha,
    float* __restrict__ partial) {
    __shared__ float al[256];
    const int rb    = (BATCH * 16 - 1) - blockIdx.x;  // reversed traversal
    const int b     = rb >> 4;
    const int chunk = rb & 15;
    const int t     = threadIdx.x;
    const int sbase = chunk * 256;

    al[t] = alpha[(size_t)b * SEQ + sbase + t];
    __syncthreads();

    const float* ep = enc + ((size_t)b * SEQ + sbase) * HID + t;
    float acc = 0.f;
#pragma unroll 4
    for (int s = 0; s < 256; ++s)
        acc += al[s] * __builtin_nontemporal_load(ep + (size_t)s * HID);

    partial[((size_t)(b * 16 + chunk)) * HID + t] = acc;
}

// ---------------------------------------------------------------------------
// Phase 3c: context[b,h] = sum_chunk partial[b,chunk,h]  (grid 64, block 256)
// ---------------------------------------------------------------------------
__global__ __launch_bounds__(256) void ta_phase3_reduce(
    const float* __restrict__ partial, float* __restrict__ out) {
    const int b = blockIdx.x, t = threadIdx.x;
    float acc = 0.f;
#pragma unroll
    for (int c = 0; c < 16; ++c)
        acc += partial[((size_t)(b * 16 + c)) * HID + t];
    out[(size_t)b * HID + t] = acc;
}

// ---------------------------------------------------------------------------
extern "C" void kernel_launch(void* const* d_in, const int* in_sizes, int n_in,
                              void* d_out, int out_size, void* d_ws, size_t ws_size,
                              hipStream_t stream) {
    (void)in_sizes; (void)n_in; (void)out_size; (void)ws_size;
    const float* enc = (const float*)d_in[0];   // [B,S,H]
    const float* fh  = (const float*)d_in[1];   // [B,H]
    const float* W1  = (const float*)d_in[2];   // [H,H]
    const float* W2  = (const float*)d_in[3];   // [H,H]
    const float* v   = (const float*)d_in[4];   // [H]
    float* out = (float*)d_out;                 // [B,H]

    // Workspace layout: W1bf (bf16, 128KB) | h2 | scores | alpha | partial
    __bf16* W1bf   = (__bf16*)d_ws;                   // 256*256 bf16
    float* fws     = (float*)((char*)d_ws + (size_t)HID * HID * sizeof(unsigned short));
    float* h2      = fws;                             // 64*256
    float* scores  = h2 + BATCH * HID;                // 64*4096
    float* alpha   = scores + (size_t)BATCH * SEQ;    // 64*4096
    float* partial = alpha + (size_t)BATCH * SEQ;     // 64*16*256

    ta_phase0_w1bf<<<(HID * HID) / 1024, 256, 0, stream>>>(W1, W1bf);
    ta_phase1_h2<<<BATCH, 256, 0, stream>>>(fh, W2, h2);
    ta_phase2_scores<<<(BATCH * (SEQ / 16)) / 8, 256, 0, stream>>>(enc, W1bf, h2, v, scores);
    ta_phase3_softmax<<<BATCH, 256, 0, stream>>>(scores, alpha);
    ta_phase3_context_partial<<<BATCH * 16, 256, 0, stream>>>(enc, alpha, partial);
    ta_phase3_reduce<<<BATCH, 256, 0, stream>>>(partial, out);
}
